// MaskedTransformerClassifier_15814069584506
// MI455X (gfx1250) — matrix operand order repaired and checked by reading, not compiled, not run on hardware
//
#include <hip/hip_runtime.h>

// ---------------------------------------------------------------------------
// Masked transformer classifier (CCT-style) for MI455X / gfx1250, wave32.
// All big matmuls run on v_wmma_f32_16x16x32_f16 (f16 in, f32 accumulate).
// Attention is single-pass flash style (online softmax, no NxN in HBM).
// GEMM global->LDS staging is software-pipelined: loads for tile k+1 are
// issued before the WMMAs of tile k, so HBM traffic overlaps matrix math.
// ---------------------------------------------------------------------------

#define Bz 2
#define Nn 1024          // P*L tokens
#define Dd 768
#define Hh 12
#define HD 64
#define FFd 3072
#define NLay 12
#define NCc 1000

typedef __attribute__((ext_vector_type(16))) _Float16 v16h;
typedef __attribute__((ext_vector_type(8)))  _Float16 v8h;
typedef __attribute__((ext_vector_type(8)))  float    v8f;

#define NEGF (-3.402823466e38f)

__device__ __forceinline__ float gelu_f(float x) {
    return 0.5f * x * (1.0f + erff(x * 0.70710678118654752f));
}

// Build a 16-half WMMA fragment from two contiguous 8-half LDS chunks.
// A-style and B-style: step = 16 (chunks at +0 and +16).
__device__ __forceinline__ v16h frag2(const _Float16* base, int step) {
    v8h lo = *(const v8h*)base;
    v8h hv = *(const v8h*)(base + step);
    v16h r;
#pragma unroll
    for (int i = 0; i < 8; ++i) { r[i] = lo[i]; r[8 + i] = hv[i]; }
    return r;
}

__device__ __forceinline__ v8f wmma_f16(v16h a, v16h b, v8f c) {
    return __builtin_amdgcn_wmma_f32_16x16x32_f16(false, a, false, b,
                                                  (short)0, c, false, false);
}

// ---------------------------------------------------------------------------
// h = x + pe  (pe broadcast over batch)
// ---------------------------------------------------------------------------
__global__ void add_pe_kernel(const float* __restrict__ x,
                              const float* __restrict__ pe,
                              float* __restrict__ h) {
    int idx = blockIdx.x * blockDim.x + threadIdx.x;
    h[idx] = x[idx] + pe[idx % (Nn * Dd)];
}

// ---------------------------------------------------------------------------
// LayerNorm over last dim (D = 768), one block (256 thr) per token.
// ---------------------------------------------------------------------------
__global__ __launch_bounds__(256)
void ln_kernel(const float* __restrict__ x, const float* __restrict__ w,
               const float* __restrict__ bb, float* __restrict__ y) {
    int tok = blockIdx.x;
    int tid = threadIdx.x;
    const float* xr = x + (size_t)tok * Dd;
    float v0 = xr[tid], v1 = xr[tid + 256], v2 = xr[tid + 512];
    __shared__ float r1[256], r2[256];
    r1[tid] = v0 + v1 + v2;
    r2[tid] = v0 * v0 + v1 * v1 + v2 * v2;
    __syncthreads();
    for (int s = 128; s > 0; s >>= 1) {
        if (tid < s) { r1[tid] += r1[tid + s]; r2[tid] += r2[tid + s]; }
        __syncthreads();
    }
    float mu  = r1[0] * (1.0f / Dd);
    float var = r2[0] * (1.0f / Dd) - mu * mu;
    float rs  = rsqrtf(var + 1e-5f);
    float* yr = y + (size_t)tok * Dd;
    yr[tid]       = (v0 - mu) * rs * w[tid]       + bb[tid];
    yr[tid + 256] = (v1 - mu) * rs * w[tid + 256] + bb[tid + 256];
    yr[tid + 512] = (v2 - mu) * rs * w[tid + 512] + bb[tid + 512];
}

// ---------------------------------------------------------------------------
// WMMA GEMM: C[M,Nout] = epilogue(A[M,K] @ W[K,Nout] + bias [+gelu] [+Res])
// Requires M%128==0, K%32==0, Nout%64==0 (true for every call site here).
// Block: 256 thr (8 waves); tile 128x64; each wave does 32x32 (2x2 WMMA).
// Software pipelined: global loads for the next K-tile are in flight while
// the WMMAs for the current tile execute.
// ---------------------------------------------------------------------------
#define LDA 40   // 32 + 8 pad, halves
#define LDB 40

__global__ __launch_bounds__(256)
void gemm_wmma_kernel(const float* __restrict__ A, const float* __restrict__ W,
                      const float* __restrict__ bias, const float* __restrict__ Res,
                      float* __restrict__ C, int M, int K, int Nout, int act_gelu) {
    __shared__ __align__(16) _Float16 sA[128 * LDA];
    __shared__ __align__(16) _Float16 sWt[64 * LDB];

    const int tid  = threadIdx.x;
    const int lane = tid & 31;
    const int wid  = tid >> 5;
    const int lm   = lane & 15;
    const int hi   = lane >> 4;
    const int m0   = blockIdx.y * 128;
    const int n0   = blockIdx.x * 64;
    const int wm   = (wid & 3) * 32;
    const int wn   = (wid >> 2) * 32;

    // Per-thread staging coordinates.
    const int ar = tid >> 3;         // A row (0..31) + p*32
    const int ac = (tid & 7) * 4;    // A col
    const int wr = tid >> 4;         // W row (0..15) + p*16
    const int wc = (tid & 15) * 4;   // W col

    v8f acc[2][2];
#pragma unroll
    for (int i = 0; i < 2; ++i)
#pragma unroll
        for (int j = 0; j < 2; ++j)
#pragma unroll
            for (int v = 0; v < 8; ++v) acc[i][j][v] = 0.0f;

    float4 ra[4], rw[2];
    // Prologue: issue all loads for k0 = 0 (kept in flight together).
#pragma unroll
    for (int p = 0; p < 4; ++p)
        ra[p] = *(const float4*)(A + (size_t)(m0 + p * 32 + ar) * K + ac);
#pragma unroll
    for (int p = 0; p < 2; ++p)
        rw[p] = *(const float4*)(W + (size_t)(p * 16 + wr) * Nout + n0 + wc);

    for (int k0 = 0; k0 < K; k0 += 32) {
        // Commit staged registers to LDS (f16, W transposed).
#pragma unroll
        for (int p = 0; p < 4; ++p) {
            _Float16* d = &sA[(p * 32 + ar) * LDA + ac];
            d[0] = (_Float16)ra[p].x; d[1] = (_Float16)ra[p].y;
            d[2] = (_Float16)ra[p].z; d[3] = (_Float16)ra[p].w;
        }
#pragma unroll
        for (int p = 0; p < 2; ++p) {
            int kr = p * 16 + wr;
            sWt[(wc + 0) * LDB + kr] = (_Float16)rw[p].x;
            sWt[(wc + 1) * LDB + kr] = (_Float16)rw[p].y;
            sWt[(wc + 2) * LDB + kr] = (_Float16)rw[p].z;
            sWt[(wc + 3) * LDB + kr] = (_Float16)rw[p].w;
        }
        __syncthreads();

        // Issue next tile's global loads; they overlap the WMMAs below.
        const int kn = k0 + 32;
        if (kn < K) {
#pragma unroll
            for (int p = 0; p < 4; ++p)
                ra[p] = *(const float4*)(A + (size_t)(m0 + p * 32 + ar) * K + kn + ac);
#pragma unroll
            for (int p = 0; p < 2; ++p)
                rw[p] = *(const float4*)(W + (size_t)(kn + p * 16 + wr) * Nout + n0 + wc);
            if (kn + 32 < K)  // prefetch tile k+2 into L2 (global_prefetch_b8)
                __builtin_prefetch(W + (size_t)(kn + 32 + wr) * Nout + n0 + wc, 0, 1);
        }

        v16h a0 = frag2(&sA[(wm + lm) * LDA + hi * 8], 16);
        v16h a1 = frag2(&sA[(wm + 16 + lm) * LDA + hi * 8], 16);
        v16h b0 = frag2(&sWt[(wn + lm) * LDB + hi * 8], 16);
        v16h b1 = frag2(&sWt[(wn + 16 + lm) * LDB + hi * 8], 16);

        acc[0][0] = wmma_f16(a0, b0, acc[0][0]);
        acc[0][1] = wmma_f16(a0, b1, acc[0][1]);
        acc[1][0] = wmma_f16(a1, b0, acc[1][0]);
        acc[1][1] = wmma_f16(a1, b1, acc[1][1]);
        __syncthreads();
    }

    // Epilogue: bias -> (gelu) -> (+residual) -> store fp32.
#pragma unroll
    for (int ti = 0; ti < 2; ++ti)
#pragma unroll
        for (int tj = 0; tj < 2; ++tj) {
            int gcol = n0 + wn + tj * 16 + lm;
            float bi = bias ? bias[gcol] : 0.0f;
#pragma unroll
            for (int v = 0; v < 8; ++v) {
                int grow = m0 + wm + ti * 16 + v + 8 * hi;
                float val = acc[ti][tj][v] + bi;
                if (act_gelu) val = gelu_f(val);
                if (Res) val += Res[(size_t)grow * Nout + gcol];
                C[(size_t)grow * Nout + gcol] = val;
            }
        }
}

// ---------------------------------------------------------------------------
// Flash attention, one (b, h, 64-query-row) tile per block; 4 waves, each
// wave owns 16 query rows. qkv layout: (B,N,3,H,HD) flattened.
// o layout: (B,N,D) with head offset h*HD (matches transpose+reshape).
// ---------------------------------------------------------------------------
#define AROW 72   // padded row length (halves) for 64-wide tiles

__global__ __launch_bounds__(128)
void attn_kernel(const float* __restrict__ qkv,
                 const unsigned char* __restrict__ mask,
                 float* __restrict__ o) {
    __shared__ __align__(16) _Float16 sq[64 * AROW];
    __shared__ __align__(16) _Float16 sk[64 * AROW];
    __shared__ __align__(16) _Float16 svt[64 * AROW];  // V transposed: [d][key]
    __shared__ __align__(16) _Float16 sp[64 * AROW];   // P tiles, per-wave rows

    const int tid  = threadIdx.x;
    const int lane = tid & 31;
    const int w    = tid >> 5;
    const int lm   = lane & 15;
    const int hi   = lane >> 4;
    const int qt0  = blockIdx.x * 64;
    const int hIdx = blockIdx.y;
    const int b    = blockIdx.z;
    const size_t rowStride = 3 * Dd;  // 2304

    const int sr = tid >> 4;         // staging row (0..7) + p*8
    const int sc = (tid & 15) * 4;   // staging col

    // Load Q tile (64 rows x 64 d) as f16 — loads batched, then converted.
    {
        float4 rq[8];
#pragma unroll
        for (int p = 0; p < 8; ++p)
            rq[p] = *(const float4*)(qkv + (size_t)(b * Nn + qt0 + p * 8 + sr) *
                                     rowStride + hIdx * HD + sc);
#pragma unroll
        for (int p = 0; p < 8; ++p) {
            _Float16* d = &sq[(p * 8 + sr) * AROW + sc];
            d[0] = (_Float16)rq[p].x; d[1] = (_Float16)rq[p].y;
            d[2] = (_Float16)rq[p].z; d[3] = (_Float16)rq[p].w;
        }
    }
    __syncthreads();

    // Per-wave Q fragments (constant across key tiles).
    v16h aq[2];
#pragma unroll
    for (int ks = 0; ks < 2; ++ks)
        aq[ks] = frag2(&sq[(w * 16 + lm) * AROW + ks * 32 + hi * 8], 16);

    // Query-row mask bits for this lane's 8 accumulator rows.
    int mq[8];
#pragma unroll
    for (int v = 0; v < 8; ++v)
        mq[v] = (int)mask[b * Nn + qt0 + w * 16 + v + 8 * hi];

    float m_i[8], l_i[8];
    v8f acc_o[4];
#pragma unroll
    for (int v = 0; v < 8; ++v) { m_i[v] = NEGF; l_i[v] = 0.0f; }
#pragma unroll
    for (int t = 0; t < 4; ++t)
#pragma unroll
        for (int v = 0; v < 8; ++v) acc_o[t][v] = 0.0f;

    for (int kt = 0; kt < Nn / 64; ++kt) {
        // Stage K (row-major) and V (transposed) tiles as f16.
        // All 16 global_load_b128 issued before any conversion/DS store.
        {
            float4 kreg[8], vreg[8];
#pragma unroll
            for (int p = 0; p < 8; ++p) {
                size_t base = (size_t)(b * Nn + kt * 64 + p * 8 + sr) * rowStride +
                              hIdx * HD + sc;
                kreg[p] = *(const float4*)(qkv + base + Dd);
                vreg[p] = *(const float4*)(qkv + base + 2 * Dd);
            }
#pragma unroll
            for (int p = 0; p < 8; ++p) {
                int r = p * 8 + sr;
                _Float16* d = &sk[r * AROW + sc];
                d[0] = (_Float16)kreg[p].x; d[1] = (_Float16)kreg[p].y;
                d[2] = (_Float16)kreg[p].z; d[3] = (_Float16)kreg[p].w;
                svt[(sc + 0) * AROW + r] = (_Float16)vreg[p].x;
                svt[(sc + 1) * AROW + r] = (_Float16)vreg[p].y;
                svt[(sc + 2) * AROW + r] = (_Float16)vreg[p].z;
                svt[(sc + 3) * AROW + r] = (_Float16)vreg[p].w;
            }
        }
        __syncthreads();

        int mk[4];
#pragma unroll
        for (int t = 0; t < 4; ++t)
            mk[t] = (int)mask[b * Nn + kt * 64 + t * 16 + lm];

        // S = Q @ K^T for this wave's 16 rows x 64 keys (4 tiles).
        v8f s[4];
#pragma unroll
        for (int t = 0; t < 4; ++t) {
#pragma unroll
            for (int v = 0; v < 8; ++v) s[t][v] = 0.0f;
#pragma unroll
            for (int ks = 0; ks < 2; ++ks) {
                v16h bk = frag2(&sk[(t * 16 + lm) * AROW + ks * 32 + hi * 8], 16);
                s[t] = wmma_f16(aq[ks], bk, s[t]);
            }
        }

        // Scale + mask.
#pragma unroll
        for (int t = 0; t < 4; ++t)
#pragma unroll
            for (int v = 0; v < 8; ++v) {
                float val = s[t][v] * 0.125f;  // HD^-0.5
                s[t][v] = (mq[v] && mk[t]) ? val : NEGF;
            }

        // Online softmax stats per row (row = v + 8*hi; 16 lanes share a row).
#pragma unroll
        for (int v = 0; v < 8; ++v) {
            float rm = fmaxf(fmaxf(s[0][v], s[1][v]), fmaxf(s[2][v], s[3][v]));
            rm = fmaxf(rm, __shfl_xor(rm, 1, 32));
            rm = fmaxf(rm, __shfl_xor(rm, 2, 32));
            rm = fmaxf(rm, __shfl_xor(rm, 4, 32));
            rm = fmaxf(rm, __shfl_xor(rm, 8, 32));
            float nm   = fmaxf(m_i[v], rm);
            float corr = __expf(m_i[v] - nm);
            m_i[v] = nm;
            float rs = 0.0f;
#pragma unroll
            for (int t = 0; t < 4; ++t) {
                float p = __expf(s[t][v] - nm);
                s[t][v] = p;
                rs += p;
            }
            rs += __shfl_xor(rs, 1, 32);
            rs += __shfl_xor(rs, 2, 32);
            rs += __shfl_xor(rs, 4, 32);
            rs += __shfl_xor(rs, 8, 32);
            l_i[v] = l_i[v] * corr + rs;
#pragma unroll
            for (int t = 0; t < 4; ++t) acc_o[t][v] *= corr;
        }

        // Spill P (C-layout) to this wave's LDS rows, reload as A-fragments.
#pragma unroll
        for (int t = 0; t < 4; ++t)
#pragma unroll
            for (int v = 0; v < 8; ++v)
                sp[(w * 16 + v + 8 * hi) * AROW + t * 16 + lm] = (_Float16)s[t][v];

        v16h ap[2];
#pragma unroll
        for (int ks = 0; ks < 2; ++ks)
            ap[ks] = frag2(&sp[(w * 16 + lm) * AROW + ks * 32 + hi * 8], 16);

        // O += P @ V  (V^T staged, so B-fragments are contiguous).
#pragma unroll
        for (int td = 0; td < 4; ++td)
#pragma unroll
            for (int ks = 0; ks < 2; ++ks) {
                v16h bv = frag2(&svt[(td * 16 + lm) * AROW + ks * 32 + hi * 8], 16);
                acc_o[td] = wmma_f16(ap[ks], bv, acc_o[td]);
            }
        __syncthreads();
    }

    // Normalize and store (B,N,D) with head offset.
#pragma unroll
    for (int td = 0; td < 4; ++td)
#pragma unroll
        for (int v = 0; v < 8; ++v) {
            int qrow = qt0 + w * 16 + v + 8 * hi;
            o[(size_t)(b * Nn + qrow) * Dd + hIdx * HD + td * 16 + lm] =
                acc_o[td][v] / l_i[v];
        }
}

// ---------------------------------------------------------------------------
// Seq-pool: scores, softmax over tokens, weighted sum, classifier.
// ---------------------------------------------------------------------------
__global__ __launch_bounds__(256)
void score_kernel(const float* __restrict__ y, const float* __restrict__ pw,
                  const float* __restrict__ pb, float* __restrict__ sc) {
    int tid = threadIdx.x, lane = tid & 31, w = tid >> 5;
    int tok = blockIdx.x * 8 + w;  // tok < B*N
    const float* r = y + (size_t)tok * Dd;
    float acc = 0.0f;
    for (int d = lane; d < Dd; d += 32) acc += r[d] * pw[d];
    for (int off = 16; off > 0; off >>= 1) acc += __shfl_xor(acc, off, 32);
    if (lane == 0) sc[tok] = acc + pb[0];
}

__global__ __launch_bounds__(256)
void softmax_tok_kernel(const float* __restrict__ sc, float* __restrict__ wts) {
    int b = blockIdx.x, tid = threadIdx.x;
    __shared__ float red[256];
    float v[4], lm = NEGF;
#pragma unroll
    for (int i = 0; i < 4; ++i) {
        v[i] = sc[b * Nn + tid + i * 256];
        lm = fmaxf(lm, v[i]);
    }
    red[tid] = lm; __syncthreads();
    for (int s = 128; s > 0; s >>= 1) {
        if (tid < s) red[tid] = fmaxf(red[tid], red[tid + s]);
        __syncthreads();
    }
    float mx = red[0]; __syncthreads();
    float ls = 0.0f;
#pragma unroll
    for (int i = 0; i < 4; ++i) { v[i] = __expf(v[i] - mx); ls += v[i]; }
    red[tid] = ls; __syncthreads();
    for (int s = 128; s > 0; s >>= 1) {
        if (tid < s) red[tid] += red[tid + s];
        __syncthreads();
    }
    float inv = 1.0f / red[0];
#pragma unroll
    for (int i = 0; i < 4; ++i) wts[b * Nn + tid + i * 256] = v[i] * inv;
}

__global__ void pooled_kernel(const float* __restrict__ wts,
                              const float* __restrict__ y,
                              float* __restrict__ pooled) {
    int idx = blockIdx.x * blockDim.x + threadIdx.x;  // over B*D
    int b = idx / Dd, d = idx % Dd;
    float acc = 0.0f;
    for (int n = 0; n < Nn; ++n)
        acc += wts[b * Nn + n] * y[(size_t)(b * Nn + n) * Dd + d];
    pooled[idx] = acc;
}

__global__ void fc_kernel(const float* __restrict__ pooled,
                          const float* __restrict__ fcw,
                          const float* __restrict__ fcb,
                          float* __restrict__ out) {
    int idx = blockIdx.x * blockDim.x + threadIdx.x;  // over B*NC
    if (idx >= Bz * NCc) return;
    int b = idx / NCc, c = idx % NCc;
    float acc = fcb[c];
    for (int d = 0; d < Dd; ++d) acc += pooled[b * Dd + d] * fcw[d * NCc + c];
    out[idx] = acc;
}

// ---------------------------------------------------------------------------
extern "C" void kernel_launch(void* const* d_in, const int* in_sizes, int n_in,
                              void* d_out, int out_size, void* d_ws, size_t ws_size,
                              hipStream_t stream) {
    (void)in_sizes; (void)n_in; (void)out_size; (void)ws_size;

    const float*         x      = (const float*)d_in[0];
    const unsigned char* mask   = (const unsigned char*)d_in[1];
    const float*         pe     = (const float*)d_in[2];
    const float*         ln0_w  = (const float*)d_in[3];
    const float*         ln0_b  = (const float*)d_in[4];
    const float*         qkv_w  = (const float*)d_in[5];
    const float*         proj_w = (const float*)d_in[6];
    const float*         proj_b = (const float*)d_in[7];
    const float*         ln1_w  = (const float*)d_in[8];
    const float*         ln1_b  = (const float*)d_in[9];
    const float*         ff1_w  = (const float*)d_in[10];
    const float*         ff1_b  = (const float*)d_in[11];
    const float*         ff2_w  = (const float*)d_in[12];
    const float*         ff2_b  = (const float*)d_in[13];
    const float*         norm_w = (const float*)d_in[14];
    const float*         norm_b = (const float*)d_in[15];
    const float*         pool_w = (const float*)d_in[16];
    const float*         pool_b = (const float*)d_in[17];
    const float*         fc_w   = (const float*)d_in[18];
    const float*         fc_b   = (const float*)d_in[19];

    // Workspace layout (fp32), ~70 MB total.
    const size_t SZ = (size_t)Bz * Nn * Dd;
    float* ws     = (float*)d_ws;
    float* hbuf   = ws;                                   // (B,N,D)
    float* ybuf   = hbuf + SZ;                            // (B,N,D)
    float* tbuf   = ybuf + SZ;                            // (B,N,D)
    float* obuf   = tbuf + SZ;                            // (B,N,D)
    float* qkvb   = obuf + SZ;                            // (B,N,3D)
    float* ffb    = qkvb + (size_t)Bz * Nn * 3 * Dd;      // (B,N,FF)
    float* scb    = ffb + (size_t)Bz * Nn * FFd;          // (B,N)
    float* wtb    = scb + Bz * Nn;                        // (B,N)
    float* poolb  = wtb + Bz * Nn;                        // (B,D)

    const int M = Bz * Nn;  // 2048

    add_pe_kernel<<<(Bz * Nn * Dd) / 256, 256, 0, stream>>>(x, pe, hbuf);

    float* h = hbuf;
    float* t = tbuf;
    for (int l = 0; l < NLay; ++l) {
        const float* qw  = qkv_w  + (size_t)l * Dd * 3 * Dd;
        const float* pw  = proj_w + (size_t)l * Dd * Dd;
        const float* pb  = proj_b + (size_t)l * Dd;
        const float* f1w = ff1_w  + (size_t)l * Dd * FFd;
        const float* f1b = ff1_b  + (size_t)l * FFd;
        const float* f2w = ff2_w  + (size_t)l * FFd * Dd;
        const float* f2b = ff2_b  + (size_t)l * Dd;

        // y = LN(h)
        ln_kernel<<<M, 256, 0, stream>>>(h, ln0_w + l * Dd, ln0_b + l * Dd, ybuf);
        // qkv = y @ qw        (no bias)
        gemm_wmma_kernel<<<dim3(3 * Dd / 64, M / 128), 256, 0, stream>>>(
            ybuf, qw, nullptr, nullptr, qkvb, M, Dd, 3 * Dd, 0);
        // o = attention(qkv, mask)
        attn_kernel<<<dim3(Nn / 64, Hh, Bz), 128, 0, stream>>>(qkvb, mask, obuf);
        // t = h + o @ pw + pb
        gemm_wmma_kernel<<<dim3(Dd / 64, M / 128), 256, 0, stream>>>(
            obuf, pw, pb, h, t, M, Dd, Dd, 0);
        // h = LN(t)
        ln_kernel<<<M, 256, 0, stream>>>(t, ln1_w + l * Dd, ln1_b + l * Dd, h);
        // ffb = gelu(h @ f1w + f1b)
        gemm_wmma_kernel<<<dim3(FFd / 64, M / 128), 256, 0, stream>>>(
            h, f1w, f1b, nullptr, ffb, M, Dd, FFd, 1);
        // t = h + ffb @ f2w + f2b
        gemm_wmma_kernel<<<dim3(Dd / 64, M / 128), 256, 0, stream>>>(
            ffb, f2w, f2b, h, t, M, FFd, Dd, 0);
        // h <- t
        float* tmp = h; h = t; t = tmp;
    }

    // Final LN, seq-pool softmax, classifier.
    ln_kernel<<<M, 256, 0, stream>>>(h, norm_w, norm_b, ybuf);
    score_kernel<<<M / 8, 256, 0, stream>>>(ybuf, pool_w, pool_b, scb);
    softmax_tok_kernel<<<Bz, 256, 0, stream>>>(scb, wtb);
    pooled_kernel<<<(Bz * Dd) / 256, 256, 0, stream>>>(wtb, ybuf, poolb);
    fc_kernel<<<(Bz * NCc + 255) / 256, 256, 0, stream>>>(poolb, fc_w, fc_b,
                                                          (float*)d_out);
}
